// Backbone_33337536152106
// MI455X (gfx1250) — matrix operand order repaired and checked by reading, not compiled, not run on hardware
//
#include <hip/hip_runtime.h>
#include <math.h>

// ---- problem constants (match reference) ----
#define H_   128
#define NH_  8
#define DH_  16
#define A_   64
#define T_   50
#define KM_  6
#define M_   19200      // A*T*K
#define AT_  3200
#define L_   512
#define FF_  512
#define E_   150000
#define AK_  384        // A*K
#define F2_  120        // 2*F

typedef __attribute__((ext_vector_type(16))) _Float16 v16h;
typedef __attribute__((ext_vector_type(8)))  float    v8f;

// =================== WMMA GEMM ===================
// C[M,N] = act(A[M,K] @ B[K,N] + bias) + resid ; fp32 in memory, f16 tiles, f32 accum.
// Block tile 64x128, BK=32, 256 threads = 8 wave32, each wave a 32x32 tile (2x2 WMMA).
#define BM_ 64
#define BN_ 128
#define BK_ 32
#define LDSP 40   // padded row stride (halfs) -> 80B, conflict-friendly

__global__ __launch_bounds__(256)
void gemm_wmma_f16(const float* __restrict__ Ag, const float* __restrict__ Bg,
                   const float* __restrict__ bias, const float* __restrict__ resid,
                   float* __restrict__ Cg, int Mr, int N, int K, int relu)
{
    __shared__ __align__(16) _Float16 As[BM_][LDSP];
    __shared__ __align__(16) _Float16 Bs[BN_][LDSP];   // transposed: [n][k]

    const int tid  = threadIdx.x;
    const int wave = tid >> 5, lane = tid & 31;
    const int wm   = wave >> 2, wn = wave & 3;        // 2x4 wave grid
    const int lrow = lane & 15, lhi = lane >> 4;
    const int m0   = blockIdx.x * BM_;
    const int n0   = blockIdx.y * BN_;

    // uniform per-block conditions (scalar branches, no lane divergence)
    const bool full_mn = (m0 + BM_ <= Mr) && (n0 + BN_ <= N) && ((K & 3) == 0) && ((N & 3) == 0);

    v8f acc[2][2] = {};

    for (int k0 = 0; k0 < K; k0 += BK_) {
        if (full_mn && (k0 + BK_ <= K)) {
            // ---- fast staging: vectorized, no predication ----
            // A tile: 64x32 floats = 512 float4 loads -> packed 4xf16 ds_store_b64
#pragma unroll
            for (int it = 0; it < (BM_ * BK_) / 4 / 256; ++it) {
                int idx = tid + it * 256;
                int r = idx >> 3;            // 8 float4 per row
                int c = (idx & 7) * 4;
                float4 v = *(const float4*)&Ag[(size_t)(m0 + r) * K + (k0 + c)];
                union { _Float16 h[4]; double d; } pk;
                pk.h[0] = (_Float16)v.x; pk.h[1] = (_Float16)v.y;
                pk.h[2] = (_Float16)v.z; pk.h[3] = (_Float16)v.w;
                *(double*)&As[r][c] = pk.d;
            }
            // B tile: 32x128 floats = 1024 float4 loads -> transposed b16 stores
#pragma unroll
            for (int it = 0; it < (BN_ * BK_) / 4 / 256; ++it) {
                int idx = tid + it * 256;
                int kk = idx >> 5;           // 32 float4 per k-row
                int nn = (idx & 31) * 4;
                float4 v = *(const float4*)&Bg[(size_t)(k0 + kk) * N + (n0 + nn)];
                Bs[nn + 0][kk] = (_Float16)v.x;
                Bs[nn + 1][kk] = (_Float16)v.y;
                Bs[nn + 2][kk] = (_Float16)v.z;
                Bs[nn + 3][kk] = (_Float16)v.w;
            }
        } else {
            // ---- tail staging: branch-free clamped loads + select ----
            for (int idx = tid; idx < BM_ * BK_; idx += 256) {
                int r = idx >> 5, c = idx & 31;
                int gr = m0 + r, gc = k0 + c;
                bool ok = (gr < Mr) && (gc < K);
                size_t off = (size_t)(ok ? gr : 0) * K + (ok ? gc : 0);
                float v = Ag[off];
                As[r][c] = (_Float16)(ok ? v : 0.0f);
            }
            for (int idx = tid; idx < BN_ * BK_; idx += 256) {
                int kk = idx >> 7, nn = idx & 127;
                int gk = k0 + kk, gn = n0 + nn;
                bool ok = (gk < K) && (gn < N);
                size_t off = (size_t)(ok ? gk : 0) * N + (ok ? gn : 0);
                float v = Bg[off];
                Bs[nn][kk] = (_Float16)(ok ? v : 0.0f);
            }
        }
        __syncthreads();

        // fragment loads (ISA 16-bit A 16x32 / B 32x16 wave32 layouts)
        union V16 { v16h v; float4 q[2]; } af[2], bf[2];
#pragma unroll
        for (int t = 0; t < 2; ++t) {
            int r = wm * 32 + t * 16 + lrow;
            af[t].q[0] = *(const float4*)&As[r][lhi * 8];        // K = lhi*8 .. +7
            af[t].q[1] = *(const float4*)&As[r][16 + lhi * 8];   // K = 16+lhi*8 .. +7
            int c = wn * 32 + t * 16 + lrow;
            bf[t].q[0] = *(const float4*)&Bs[c][lhi * 8];        // K = lhi*8 .. +7
            bf[t].q[1] = *(const float4*)&Bs[c][16 + lhi * 8];   // K = 16+lhi*8 .. +7
        }
#pragma unroll
        for (int tm = 0; tm < 2; ++tm)
#pragma unroll
            for (int tn = 0; tn < 2; ++tn)
                acc[tm][tn] = __builtin_amdgcn_wmma_f32_16x16x32_f16(
                    false, af[tm].v, false, bf[tn].v, (short)0, acc[tm][tn], false, false);
        __syncthreads();
    }

    // epilogue: C/D 16x16 f32 layout: VGPR r -> row r (lanes 0-15) / r+8 (lanes 16-31)
#pragma unroll
    for (int tm = 0; tm < 2; ++tm)
#pragma unroll
        for (int tn = 0; tn < 2; ++tn) {
            int gcol = n0 + wn * 32 + tn * 16 + lrow;
            float bv = (bias && gcol < N) ? bias[gcol] : 0.0f;
#pragma unroll
            for (int r = 0; r < 8; ++r) {
                int grow = m0 + wm * 32 + tm * 16 + lhi * 8 + r;
                if (grow < Mr && gcol < N) {
                    float v = acc[tm][tn][r] + bv;
                    if (relu)  v = fmaxf(v, 0.0f);
                    if (resid) v += resid[(size_t)grow * N + gcol];
                    Cg[(size_t)grow * N + gcol] = v;
                }
            }
        }
}

// =================== LayerNorm (H=128, one row per block) ===================
__global__ void ln_kernel(const float* __restrict__ x, const float* __restrict__ g,
                          const float* __restrict__ b, float* __restrict__ y)
{
    __shared__ float red[H_];
    int row = blockIdx.x, h = threadIdx.x;
    float v = x[(size_t)row * H_ + h];
    red[h] = v; __syncthreads();
    for (int s = 64; s > 0; s >>= 1) { if (h < s) red[h] += red[h + s]; __syncthreads(); }
    float mu = red[0] * (1.0f / H_);
    __syncthreads();
    float d = v - mu;
    red[h] = d * d; __syncthreads();
    for (int s = 64; s > 0; s >>= 1) { if (h < s) red[h] += red[h + s]; __syncthreads(); }
    float var = red[0] * (1.0f / H_);
    y[(size_t)row * H_ + h] = d * rsqrtf(var + 1e-5f) * g[h] + b[h];
}

// =================== small-K embed MLP hidden: relu(x[rows,d] @ W1[d,128] + b1) ===================
__global__ void small_mlp_hidden(const float* __restrict__ x, int d,
                                 const float* __restrict__ W1, const float* __restrict__ b1,
                                 float* __restrict__ out, int rows)
{
    long long i = (long long)blockIdx.x * blockDim.x + threadIdx.x;
    long long n = (long long)rows * H_;
    if (i >= n) return;
    int r = (int)(i >> 7), h = (int)(i & (H_ - 1));
    float s = b1[h];
    for (int j = 0; j < d; ++j) s += x[(size_t)r * d + j] * W1[j * H_ + h];
    out[i] = fmaxf(s, 0.0f);
}

// =================== elementwise helpers ===================
__global__ void fill_f32(float* __restrict__ p, float v, long long n)
{ long long i = (long long)blockIdx.x * blockDim.x + threadIdx.x; if (i < n) p[i] = v; }

__global__ void fill_u32(unsigned* __restrict__ p, unsigned v, long long n)
{ long long i = (long long)blockIdx.x * blockDim.x + threadIdx.x; if (i < n) p[i] = v; }

__global__ void bcast_mode(const float* __restrict__ mt, float* __restrict__ m, long long n)
{
    long long i = (long long)blockIdx.x * blockDim.x + threadIdx.x; if (i >= n) return;
    long long node = i >> 7; int k = (int)(node % KM_);
    m[i] = mt[k * H_ + (int)(i & (H_ - 1))];
}

// out[e,:] (+)= rowsrc[src[e],:]
__global__ void gather_rows(float* __restrict__ outp, const float* __restrict__ rowsrc,
                            const int* __restrict__ sidx, long long n, int addmode)
{
    long long i = (long long)blockIdx.x * blockDim.x + threadIdx.x; if (i >= n) return;
    int e = (int)(i >> 7), h = (int)(i & (H_ - 1));
    float v = rowsrc[(size_t)sidx[e] * H_ + h];
    outp[i] = addmode ? outp[i] + v : v;
}

// scores[e,nh] = (q[dst[e],nh*16: ] . k[e,nh*16: ]) / 4
__global__ void scores_k(const float* __restrict__ qb, const float* __restrict__ ke,
                         const int* __restrict__ dstI, float* __restrict__ sc, long long n)
{
    long long i = (long long)blockIdx.x * blockDim.x + threadIdx.x; if (i >= n) return;
    int e = (int)(i >> 3), hd = (int)(i & 7);
    const float* qp = qb + (size_t)dstI[e] * H_ + hd * DH_;
    const float* kp = ke + (size_t)e * H_ + hd * DH_;
    float s = 0.f;
#pragma unroll
    for (int j = 0; j < DH_; ++j) s += qp[j] * kp[j];
    sc[i] = s * 0.25f;   // 1/sqrt(16)
}

// segment max via order-preserving uint mapping + atomicMax
__global__ void segmax_k(const float* __restrict__ sc, const int* __restrict__ dstI,
                         unsigned* __restrict__ smax, long long n)
{
    long long i = (long long)blockIdx.x * blockDim.x + threadIdx.x; if (i >= n) return;
    unsigned u = __float_as_uint(sc[i]);
    u = (u & 0x80000000u) ? ~u : (u | 0x80000000u);
    atomicMax(&smax[(size_t)dstI[i >> 3] * NH_ + (int)(i & 7)], u);
}

__global__ void smax_fin_k(const unsigned* __restrict__ su, float* __restrict__ sf, long long n)
{
    long long i = (long long)blockIdx.x * blockDim.x + threadIdx.x; if (i >= n) return;
    unsigned u = su[i];
    unsigned bits = (u & 0x80000000u) ? (u ^ 0x80000000u) : ~u;
    float f = __uint_as_float(bits);
    if (!(fabsf(f) <= 3.0e38f)) f = 0.0f;   // non-finite (empty segment) -> 0, like reference
    sf[i] = f;
}

__global__ void expden_k(float* __restrict__ sc, const int* __restrict__ dstI,
                         const float* __restrict__ smaxf, float* __restrict__ den, long long n)
{
    long long i = (long long)blockIdx.x * blockDim.x + threadIdx.x; if (i >= n) return;
    int e = (int)(i >> 3), hd = (int)(i & 7);
    float ex = expf(sc[i] - smaxf[(size_t)dstI[e] * NH_ + hd]);
    sc[i] = ex;
    atomicAdd(&den[(size_t)dstI[e] * NH_ + hd], ex);
}

__global__ void agg_k(const float* __restrict__ sc, const float* __restrict__ den,
                      const float* __restrict__ ve, const int* __restrict__ dstI,
                      float* __restrict__ aggp, long long n)
{
    long long i = (long long)blockIdx.x * blockDim.x + threadIdx.x; if (i >= n) return;
    int e = (int)(i >> 7), h = (int)(i & (H_ - 1)), hd = h >> 4;
    int dd = dstI[e];
    float w = sc[(size_t)e * NH_ + hd] / (den[(size_t)dd * NH_ + hd] + 1e-9f);
    atomicAdd(&aggp[(size_t)dd * H_ + h], w * ve[i]);
}

__global__ void extract_last(const float* __restrict__ m, float* __restrict__ ml, long long n)
{
    long long i = (long long)blockIdx.x * blockDim.x + threadIdx.x; if (i >= n) return;
    int h = (int)(i & (H_ - 1)); int ak = (int)(i >> 7);
    int a = ak / KM_, k = ak % KM_;
    long long node = ((long long)a * T_ + (T_ - 1)) * KM_ + k;
    ml[i] = m[(size_t)node * H_ + h];
}

__global__ void add_anchor(float* __restrict__ m, const float* __restrict__ anch, long long n)
{
    long long i = (long long)blockIdx.x * blockDim.x + threadIdx.x; if (i >= n) return;
    int h = (int)(i & (H_ - 1));
    long long node = i >> 7;
    int k = (int)(node % KM_);
    int a = (int)((node / KM_) / T_);
    m[i] += anch[((size_t)a * KM_ + k) * H_ + h];
}

// =================== host orchestration ===================
static inline unsigned blks(long long n, int b) { return (unsigned)((n + b - 1) / b); }

extern "C" void kernel_launch(void* const* d_in, const int* in_sizes, int n_in,
                              void* d_out, int out_size, void* d_ws, size_t ws_size,
                              hipStream_t stream)
{
    (void)in_sizes; (void)n_in; (void)out_size; (void)ws_size;
    const float* a_input  = (const float*)d_in[0];
    const float* l_embs   = (const float*)d_in[1];
    const float* mode_tok = (const float*)d_in[2];
    const float* embW1 = (const float*)d_in[3];
    const float* embb1 = (const float*)d_in[4];
    const float* embW2 = (const float*)d_in[5];
    const float* embb2 = (const float*)d_in[6];
    const float* Wq  = (const float*)d_in[7];
    const float* Wk  = (const float*)d_in[8];
    const float* Wv  = (const float*)d_in[9];
    const float* Wo  = (const float*)d_in[10];
    const float* Wke = (const float*)d_in[11];
    const float* Wve = (const float*)d_in[12];
    const float* g1  = (const float*)d_in[13];
    const float* b1  = (const float*)d_in[14];
    const float* g2  = (const float*)d_in[15];
    const float* b2  = (const float*)d_in[16];
    const float* fW1 = (const float*)d_in[17];
    const float* fb1 = (const float*)d_in[18];
    const float* fW2 = (const float*)d_in[19];
    const float* fb2 = (const float*)d_in[20];
    const float* tp_W1 = (const float*)d_in[21];
    const float* tp_b1 = (const float*)d_in[22];
    const float* tp_W2 = (const float*)d_in[23];
    const float* tp_b2 = (const float*)d_in[24];
    const float* pa_W1 = (const float*)d_in[25];
    const float* pa_b1 = (const float*)d_in[26];
    const float* pa_W2 = (const float*)d_in[27];
    const float* pa_b2 = (const float*)d_in[28];
    const float* tr_W1 = (const float*)d_in[29];
    const float* tr_b1 = (const float*)d_in[30];
    const float* tr_W2 = (const float*)d_in[31];
    const float* tr_b2 = (const float*)d_in[32];
    const float* t2m_attr   = (const float*)d_in[33];
    const float* l2m_attr   = (const float*)d_in[34];
    const float* m2m_h_attr = (const float*)d_in[35];
    const float* m2m_a_attr = (const float*)d_in[36];
    const int* t2m_ei   = (const int*)d_in[37];
    const int* l2m_ei   = (const int*)d_in[38];
    const int* m2m_h_ei = (const int*)d_in[39];
    const int* m2m_a_ei = (const int*)d_in[40];
    const int* m2m_s_ei = (const int*)d_in[41];
    float* out = (float*)d_out;

    // ---- workspace carve ----
    float* w = (float*)d_ws;
    auto alloc = [&](size_t n) { float* p = w; w += n; return p; };
    float* m_    = alloc((size_t)M_ * H_);
    float* a_e   = alloc((size_t)AT_ * H_);
    float* xd    = alloc((size_t)M_ * H_);
    float* xs    = alloc((size_t)M_ * H_);
    float* qb    = alloc((size_t)M_ * H_);
    float* kx    = alloc((size_t)M_ * H_);
    float* vx    = alloc((size_t)M_ * H_);
    float* aggb  = alloc((size_t)M_ * H_);
    float* hlnb  = alloc((size_t)M_ * H_);
    float* xt    = alloc((size_t)M_ * H_);
    float* ffh   = alloc((size_t)M_ * FF_);
    float* keb   = alloc((size_t)E_ * H_);   // doubles as edge-embed hidden
    float* ebuf  = alloc((size_t)E_ * H_);
    float* veb   = alloc((size_t)E_ * H_);
    float* sc    = alloc((size_t)E_ * NH_);
    float* smaxb = alloc((size_t)M_ * NH_);
    float* denb  = alloc((size_t)M_ * NH_);
    float* ml    = alloc((size_t)AK_ * H_);
    float* th    = alloc((size_t)AK_ * H_);
    float* anch  = alloc((size_t)AK_ * H_);

    auto gemm = [&](const float* Ag, const float* Bg, const float* bias, const float* resid,
                    float* Cg, int Mr, int N, int Kd, int relu) {
        dim3 g(blks(Mr, BM_), blks(N, BN_));
        gemm_wmma_f16<<<g, 256, 0, stream>>>(Ag, Bg, bias, resid, Cg, Mr, N, Kd, relu);
    };

    auto attn = [&](int i, const float* xsrc, int ns, const float* attr, int d,
                    int eidx, const int* ei) {
        const int* srcI = ei;
        const int* dstI = ei + E_;
        const long long nEH = (long long)E_ * H_;
        const long long nS  = (long long)E_ * NH_;
        const long long nMN = (long long)M_ * NH_;

        ln_kernel<<<M_, H_, 0, stream>>>(m_, g1 + i * H_, b1 + i * H_, xd);
        const float* xsl = xd;
        if (xsrc != m_) {
            ln_kernel<<<ns, H_, 0, stream>>>(xsrc, g1 + i * H_, b1 + i * H_, xs);
            xsl = xs;
        }
        gemm(xd,  Wq + (size_t)i * H_ * H_, nullptr, nullptr, qb, M_, H_, H_, 0);
        gemm(xsl, Wk + (size_t)i * H_ * H_, nullptr, nullptr, kx, ns, H_, H_, 0);
        gemm(xsl, Wv + (size_t)i * H_ * H_, nullptr, nullptr, vx, ns, H_, H_, 0);

        if (attr) {
            small_mlp_hidden<<<blks(nEH, 256), 256, 0, stream>>>(
                attr, d, embW1 + (size_t)eidx * 5 * H_, embb1 + (size_t)eidx * H_, keb, E_);
            gemm(keb,  embW2 + (size_t)eidx * H_ * H_, embb2 + (size_t)eidx * H_, nullptr,
                 ebuf, E_, H_, H_, 0);
            gemm(ebuf, Wke + (size_t)i * H_ * H_, nullptr, nullptr, keb, E_, H_, H_, 0);
            gemm(ebuf, Wve + (size_t)i * H_ * H_, nullptr, nullptr, veb, E_, H_, H_, 0);
            gather_rows<<<blks(nEH, 256), 256, 0, stream>>>(keb, kx, srcI, nEH, 1);
            gather_rows<<<blks(nEH, 256), 256, 0, stream>>>(veb, vx, srcI, nEH, 1);
        } else {
            gather_rows<<<blks(nEH, 256), 256, 0, stream>>>(keb, kx, srcI, nEH, 0);
            gather_rows<<<blks(nEH, 256), 256, 0, stream>>>(veb, vx, srcI, nEH, 0);
        }

        fill_u32<<<blks(nMN, 256), 256, 0, stream>>>((unsigned*)smaxb, 0x007FFFFFu, nMN); // mapped(-inf)
        scores_k<<<blks(nS, 256), 256, 0, stream>>>(qb, keb, dstI, sc, nS);
        segmax_k<<<blks(nS, 256), 256, 0, stream>>>(sc, dstI, (unsigned*)smaxb, nS);
        smax_fin_k<<<blks(nMN, 256), 256, 0, stream>>>((const unsigned*)smaxb, smaxb, nMN);
        fill_f32<<<blks(nMN, 256), 256, 0, stream>>>(denb, 0.0f, nMN);
        expden_k<<<blks(nS, 256), 256, 0, stream>>>(sc, dstI, smaxb, denb, nS);
        fill_f32<<<blks((long long)M_ * H_, 256), 256, 0, stream>>>(aggb, 0.0f, (long long)M_ * H_);
        agg_k<<<blks(nEH, 256), 256, 0, stream>>>(sc, denb, veb, dstI, aggb, nEH);

        gemm(aggb, Wo + (size_t)i * H_ * H_, nullptr, m_, xt, M_, H_, H_, 0);      // x = m + agg@Wo
        ln_kernel<<<M_, H_, 0, stream>>>(xt, g2 + i * H_, b2 + i * H_, hlnb);
        gemm(hlnb, fW1 + (size_t)i * H_ * FF_, fb1 + (size_t)i * FF_, nullptr, ffh, M_, FF_, H_, 1);
        gemm(ffh,  fW2 + (size_t)i * FF_ * H_, fb2 + (size_t)i * H_, xt, m_, M_, H_, FF_, 0);
    };

    // ---- agent embeddings + mode-token init ----
    small_mlp_hidden<<<blks((long long)AT_ * H_, 256), 256, 0, stream>>>(
        a_input, 5, embW1, embb1, hlnb, AT_);
    gemm(hlnb, embW2, embb2, nullptr, a_e, AT_, H_, H_, 0);
    bcast_mode<<<blks((long long)M_ * H_, 256), 256, 0, stream>>>(mode_tok, m_, (long long)M_ * H_);

    // ---- proposal stage ----
    attn(0, a_e,    AT_, t2m_attr,   4, 1, t2m_ei);
    attn(1, l_embs, L_,  l2m_attr,   3, 2, l2m_ei);
    attn(2, m_,     M_,  m2m_h_attr, 4, 3, m2m_h_ei);
    attn(3, m_,     M_,  m2m_a_attr, 3, 4, m2m_a_ei);
    attn(4, m_,     M_,  nullptr,    0, 0, m2m_s_ei);

    const long long nAKH = (long long)AK_ * H_;
    extract_last<<<blks(nAKH, 256), 256, 0, stream>>>(m_, ml, nAKH);
    gemm(ml,  tp_W1, tp_b1, nullptr, th, AK_, H_, H_, 1);
    gemm(th,  tp_W2, tp_b2, nullptr, out, AK_, F2_, H_, 0);              // proposal -> d_out[0:]
    gemm(out, pa_W1, pa_b1, nullptr, th, AK_, H_, F2_, 1);
    gemm(th,  pa_W2, pa_b2, nullptr, anch, AK_, H_, H_, 0);              // anchor
    add_anchor<<<blks((long long)M_ * H_, 256), 256, 0, stream>>>(m_, anch, (long long)M_ * H_);

    // ---- refinement stage ----
    attn(5, a_e,    AT_, t2m_attr,   4, 1, t2m_ei);
    attn(6, l_embs, L_,  l2m_attr,   3, 2, l2m_ei);
    attn(7, m_,     M_,  m2m_h_attr, 4, 3, m2m_h_ei);
    attn(8, m_,     M_,  m2m_a_attr, 3, 4, m2m_a_ei);
    attn(9, m_,     M_,  nullptr,    0, 0, m2m_s_ei);

    extract_last<<<blks(nAKH, 256), 256, 0, stream>>>(m_, ml, nAKH);
    gemm(ml, tr_W1, tr_b1, nullptr, th, AK_, H_, H_, 1);
    gemm(th, tr_W2, tr_b2, out, out + (size_t)AK_ * F2_, AK_, F2_, H_, 0); // refine = proposal + head
}
